// AGRNN_33535104647433
// MI455X (gfx1250) — matrix operand order repaired and checked by reading, not compiled, not run on hardware
//
#include <hip/hip_runtime.h>
#include <hip/hip_bf16.h>

typedef __bf16 bf16_t;
typedef __attribute__((ext_vector_type(16))) __bf16 v16bf;
typedef __attribute__((ext_vector_type(8)))  __bf16 v8bf;
typedef __attribute__((ext_vector_type(8)))  float  v8f;

#define NN    16384
#define EE    131072
#define ERO   32768
#define DD    1024
#define DLP   320      /* 300 padded to 320 (5*64) */
#define DSP   64       /* 16 padded to 64 so all segment boundaries are 64-aligned */
#define CC    117
#define CPAD  128

#define KE    2112     /* 1024+1024+64 */
#define KL    640      /* 320+320 */
#define KNU   2048
#define KNL   640
#define KP    2752     /* 1024+320+64+320+1024 */

__device__ __forceinline__ bf16_t f2bf(float f) {
    union { float f; unsigned u; } in; in.f = f;
    unsigned u = in.u;
    u += 0x7fffu + ((u >> 16) & 1u);
    union { unsigned short s; bf16_t b; } out; out.s = (unsigned short)(u >> 16);
    return out.b;
}

// ---------- conversion kernels ----------
__global__ void k_cvt_rows(const float* __restrict__ src, bf16_t* __restrict__ dst,
                           int M, int Ks, int Kd) {
    unsigned idx = blockIdx.x * 256u + threadIdx.x;
    unsigned total = (unsigned)M * (unsigned)Kd;
    if (idx >= total) return;
    int r = idx / Kd, k = idx % Kd;
    float v = (k < Ks) ? src[(size_t)r * Ks + k] : 0.0f;
    dst[idx] = f2bf(v);
}

// mode 0: identity up to Ktot; mode 1: [300|pad20|300|pad20];
// mode 2: predictor concat map for K layout 1024|320|64|320|1024
__global__ void k_cvt_w(const float* __restrict__ W, bf16_t* __restrict__ Wt,
                        int Ncols, int Npad, int Kpad, int Ktot, int mode) {
    unsigned idx = blockIdx.x * 256u + threadIdx.x;
    unsigned total = (unsigned)Npad * (unsigned)Kpad;
    if (idx >= total) return;
    int n = idx / Kpad, k = idx % Kpad;
    int ks = -1;
    if (mode == 0) {
        ks = (k < Ktot) ? k : -1;
    } else if (mode == 1) {
        if (k < 300) ks = k;
        else if (k < 320) ks = -1;
        else if (k < 620) ks = k - 20;
        else ks = -1;
    } else {
        if (k < 1024) ks = k;
        else if (k < 1344) { int j = k - 1024; ks = (j < 300) ? 1024 + j : -1; }
        else if (k < 1408) { int j = k - 1344; ks = (j < 16)  ? 1324 + j : -1; }
        else if (k < 1728) { int j = k - 1408; ks = (j < 300) ? 1340 + j : -1; }
        else               { ks = 1640 + (k - 1728); }
    }
    float v = (n < Ncols && ks >= 0) ? W[(size_t)ks * Ncols + n] : 0.0f;
    Wt[idx] = f2bf(v);
}

__global__ void k_counts(const int* __restrict__ dst, float* __restrict__ cnt, int E) {
    int e = blockIdx.x * 256 + threadIdx.x;
    if (e < E) unsafeAtomicAdd(&cnt[dst[e]], 1.0f);
}

__global__ void k_norm(const float* __restrict__ agg, const float* __restrict__ cnt,
                       bf16_t* __restrict__ out, int K) {
    unsigned idx = blockIdx.x * 256u + threadIdx.x;
    unsigned total = (unsigned)NN * (unsigned)K;
    if (idx >= total) return;
    int r = idx / K;
    float c = fmaxf(cnt[r], 1.0f);
    out[idx] = f2bf(agg[idx] / c);
}

// ---------- fused gather-GEMM kernel ----------
// 256 threads = 8 wave32; block tile M=256 x N=64; K staged 64 halves per barrier.
// Each wave owns 32 M-rows: per 32-K chunk = 2 A frags + 4 B frags -> 8 WMMA
// (B frag reused for both A frags from registers); 16 WMMA per stage per wave.
// G: 0 edge-app, 1 edge-lang, 2 node-app, 3 node-lang, 4 readout.
// EP: 0 relu+bias -> f32 atomic scatter by dst; 1 relu+bias -> bf16 store; 2 bias -> f32 store.
template <int G, int EP>
__global__ __launch_bounds__(256)
void gemm_ws(const bf16_t* __restrict__ p0, const bf16_t* __restrict__ p1,
             const bf16_t* __restrict__ p2, const bf16_t* __restrict__ Wt, int Kpad,
             const int* __restrict__ srcIdx, const int* __restrict__ dstIdx,
             const float* __restrict__ bias, int Nvalid,
             float* __restrict__ aggOut, bf16_t* __restrict__ bfOut,
             float* __restrict__ fOut, int outStride) {
    __shared__ bf16_t As[256 * 72];   // 64 staged halves + 8 pad per row
    __shared__ bf16_t Bs[64 * 72];
    __shared__ int dIdxS[256];

    const int tid   = threadIdx.x;
    const int mBase = blockIdx.x * 256;
    const int nBase = blockIdx.y * 64;

    // ---- staging assignment: one A row per thread; B 16-half chunk per thread ----
    const int arow  = tid;                 // A row this thread stages (0..255)
    const int bn    = tid >> 2;            // B row (output col) this thread stages
    const int bpart = (tid & 3) * 16;      // 16-half chunk within 64

    int sI = 0, dI = 0;
    if constexpr (G == 0 || G == 1 || G == 4) {
        sI = srcIdx[mBase + arow];
        dI = dstIdx[mBase + arow];
    }
    if constexpr (EP == 0) dIdxS[tid] = dI;

    const int w     = tid >> 5;
    const int lane  = tid & 31;
    const int laneM = lane & 15;
    const int hi    = lane >> 4;
    const int frow0 = w * 32 + laneM;      // wave's first 16-row A fragment
    const int frow1 = frow0 + 16;          // second 16-row A fragment
    const int aoff1 = hi ? 16 : 0;         // A frag: lanes0-15 K0..15, lanes16-31 K16..31
    const int aoff2 = 8 + (hi ? 16 : 0);
    const int boff  = hi ? 16 : 0;         // B frag: lanes0-15 K0..15, lanes16-31 K16..31

    v8f c[2][4];
    #pragma unroll
    for (int s = 0; s < 2; ++s)
        #pragma unroll
        for (int t = 0; t < 4; ++t) c[s][t] = v8f{};

    const bf16_t* bp = Wt + (size_t)(nBase + bn) * Kpad + bpart;  // advances 64/step

    auto loadFragA = [&](int row, int kk) {
        v16bf a;
        v8bf alo = *(const v8bf*)(As + row * 72 + kk + aoff1);
        v8bf ahi = *(const v8bf*)(As + row * 72 + kk + aoff2);
        #pragma unroll
        for (int i = 0; i < 8; ++i) { a[i] = alo[i]; a[i + 8] = ahi[i]; }
        return a;
    };

    // one 64-K stage: tight loop, base pointers just increment
    auto span = [&](const bf16_t* ap, int nsteps) {
        for (int s = 0; s < nsteps; ++s) {
            v8bf av[8];
            #pragma unroll
            for (int i = 0; i < 8; ++i) av[i] = *(const v8bf*)(ap + i * 8);
            v8bf g0 = *(const v8bf*)(bp);
            v8bf g1 = *(const v8bf*)(bp + 8);
            __syncthreads();               // previous stage's LDS reads done
            bf16_t* adst = As + arow * 72;
            #pragma unroll
            for (int i = 0; i < 8; ++i) *(v8bf*)(adst + i * 8) = av[i];
            bf16_t* bdst = Bs + bn * 72 + bpart;
            *(v8bf*)(bdst)     = g0;
            *(v8bf*)(bdst + 8) = g1;
            __syncthreads();
            #pragma unroll
            for (int kk = 0; kk < 64; kk += 32) {
                v16bf a0 = loadFragA(frow0, kk);
                v16bf a1 = loadFragA(frow1, kk);
                #pragma unroll
                for (int t = 0; t < 4; ++t) {
                    v16bf b;
                    const bf16_t* bsrc = Bs + (t * 16 + laneM) * 72 + kk + boff;
                    v8bf blo = *(const v8bf*)(bsrc);
                    v8bf bhi = *(const v8bf*)(bsrc + 8);
                    #pragma unroll
                    for (int i = 0; i < 8; ++i) { b[i] = blo[i]; b[i + 8] = bhi[i]; }
                    c[0][t] = __builtin_amdgcn_wmma_f32_16x16x32_bf16(false, a0, false, b,
                                                                      (short)0, c[0][t], false, false);
                    c[1][t] = __builtin_amdgcn_wmma_f32_16x16x32_bf16(false, a1, false, b,
                                                                      (short)0, c[1][t], false, false);
                }
            }
            ap += 64; bp += 64;
        }
    };

    // ---- per-gather segment spans (boundaries all multiples of 64) ----
    if constexpr (G == 0) {
        span(p0 + (size_t)sI * DD, 16);                 // feat[src]
        span(p0 + (size_t)dI * DD, 16);                 // feat[dst]
        span(p2 + (size_t)(mBase + arow) * DSP, 1);     // s_f (padded)
    } else if constexpr (G == 1) {
        span(p0 + (size_t)sI * DLP, 5);                 // w2v[src]
        span(p0 + (size_t)dI * DLP, 5);                 // w2v[dst]
    } else if constexpr (G == 2) {
        span(p0 + (size_t)(mBase + arow) * DD, 16);     // feat
        span(p1 + (size_t)(mBase + arow) * DD, 16);     // agg
    } else if constexpr (G == 3) {
        span(p0 + (size_t)(mBase + arow) * DLP, 5);     // w2v
        span(p1 + (size_t)(mBase + arow) * DLP, 5);     // aggl
    } else {
        span(p0 + (size_t)dI * DD, 16);                 // nnf[dst]
        span(p1 + (size_t)dI * DLP, 5);                 // nnfl[dst]
        span(p2 + (size_t)(mBase + arow) * DSP, 1);     // s_f_ro
        span(p1 + (size_t)sI * DLP, 5);                 // nnfl[src]
        span(p0 + (size_t)sI * DD, 16);                 // nnf[src]
    }

    // ---- epilogue: C layout lanes0-15 col=lane,M=v ; lanes16-31 col=lane-16,M=8+v ----
    #pragma unroll
    for (int s = 0; s < 2; ++s) {
        #pragma unroll
        for (int t = 0; t < 4; ++t) {
            const int col = nBase + t * 16 + laneM;
            const float bv = (col < Nvalid) ? bias[col] : 0.0f;
            #pragma unroll
            for (int v = 0; v < 8; ++v) {
                const int lrow = w * 32 + s * 16 + hi * 8 + v;
                const int gRow = mBase + lrow;
                float val = c[s][t][v] + bv;
                if constexpr (EP == 0) {
                    val = fmaxf(val, 0.0f);
                    if (col < Nvalid)
                        unsafeAtomicAdd(aggOut + (size_t)dIdxS[lrow] * outStride + col, val);
                } else if constexpr (EP == 1) {
                    val = fmaxf(val, 0.0f);
                    bfOut[(size_t)gRow * outStride + col] = f2bf(val);
                } else {
                    if (col < Nvalid)
                        fOut[(size_t)gRow * outStride + col] = val;
                }
            }
        }
    }
}

extern "C" void kernel_launch(void* const* d_in, const int* in_sizes, int n_in,
                              void* d_out, int out_size, void* d_ws, size_t ws_size,
                              hipStream_t stream) {
    const float* feat   = (const float*)d_in[0];
    const float* w2v    = (const float*)d_in[1];
    const float* s_f    = (const float*)d_in[2];
    const float* s_f_ro = (const float*)d_in[3];
    const float* W_e    = (const float*)d_in[4];
    const float* b_e    = (const float*)d_in[5];
    const float* W_el   = (const float*)d_in[6];
    const float* b_el   = (const float*)d_in[7];
    const float* W_nu   = (const float*)d_in[8];
    const float* b_nu   = (const float*)d_in[9];
    const float* W_nul  = (const float*)d_in[10];
    const float* b_nul  = (const float*)d_in[11];
    const float* W_p    = (const float*)d_in[12];
    const float* b_p    = (const float*)d_in[13];
    const int* edge_src = (const int*)d_in[14];
    const int* edge_dst = (const int*)d_in[15];
    const int* ro_src   = (const int*)d_in[16];
    const int* ro_dst   = (const int*)d_in[17];
    float* out = (float*)d_out;

    char* base = (char*)d_ws;
    size_t off = 0;
    auto carve = [&](size_t bytes) -> char* {
        size_t o = (off + 255) & ~(size_t)255;
        off = o + bytes;
        return base + o;
    };
    bf16_t* featb  = (bf16_t*)carve((size_t)NN * DD * 2);
    bf16_t* w2vb   = (bf16_t*)carve((size_t)NN * DLP * 2);
    bf16_t* sfb    = (bf16_t*)carve((size_t)EE * DSP * 2);
    bf16_t* sfrob  = (bf16_t*)carve((size_t)ERO * DSP * 2);
    bf16_t* WtE    = (bf16_t*)carve((size_t)1024 * KE * 2);
    bf16_t* WtEL   = (bf16_t*)carve((size_t)DLP * KL * 2);
    bf16_t* WtNU   = (bf16_t*)carve((size_t)1024 * KNU * 2);
    bf16_t* WtNUL  = (bf16_t*)carve((size_t)DLP * KNL * 2);
    bf16_t* WtP    = (bf16_t*)carve((size_t)CPAD * KP * 2);
    float*  agg    = (float*) carve((size_t)NN * DD * 4);
    float*  aggl   = (float*) carve((size_t)NN * DLP * 4);
    float*  cnt    = (float*) carve((size_t)NN * 4);
    bf16_t* aggb   = (bf16_t*)carve((size_t)NN * DD * 2);
    bf16_t* agglb  = (bf16_t*)carve((size_t)NN * DLP * 2);
    bf16_t* nnfb   = (bf16_t*)carve((size_t)NN * DD * 2);
    bf16_t* nnflb  = (bf16_t*)carve((size_t)NN * DLP * 2);
    (void)in_sizes; (void)n_in; (void)out_size; (void)ws_size;

    auto cdiv = [](unsigned a, unsigned b) { return (a + b - 1) / b; };

    hipMemsetAsync(agg,  0, (size_t)NN * DD * 4, stream);
    hipMemsetAsync(aggl, 0, (size_t)NN * DLP * 4, stream);
    hipMemsetAsync(cnt,  0, (size_t)NN * 4, stream);

    k_cvt_rows<<<cdiv(NN * DD, 256), 256, 0, stream>>>(feat, featb, NN, DD, DD);
    k_cvt_rows<<<cdiv(NN * DLP, 256), 256, 0, stream>>>(w2v, w2vb, NN, 300, DLP);
    k_cvt_rows<<<cdiv(EE * DSP, 256), 256, 0, stream>>>(s_f, sfb, EE, 16, DSP);
    k_cvt_rows<<<cdiv(ERO * DSP, 256), 256, 0, stream>>>(s_f_ro, sfrob, ERO, 16, DSP);
    k_cvt_w<<<cdiv(1024 * KE, 256), 256, 0, stream>>>(W_e, WtE, 1024, 1024, KE, 2064, 0);
    k_cvt_w<<<cdiv(DLP * KL, 256), 256, 0, stream>>>(W_el, WtEL, 300, DLP, KL, 0, 1);
    k_cvt_w<<<cdiv(1024 * KNU, 256), 256, 0, stream>>>(W_nu, WtNU, 1024, 1024, KNU, 2048, 0);
    k_cvt_w<<<cdiv(DLP * KNL, 256), 256, 0, stream>>>(W_nul, WtNUL, 300, DLP, KNL, 0, 1);
    k_cvt_w<<<cdiv(CPAD * KP, 256), 256, 0, stream>>>(W_p, WtP, CC, CPAD, KP, 0, 2);

    k_counts<<<cdiv(EE, 256), 256, 0, stream>>>(edge_dst, cnt, EE);

    gemm_ws<0, 0><<<dim3(EE / 256, DD / 64), 256, 0, stream>>>(
        featb, nullptr, sfb, WtE, KE, edge_src, edge_dst, b_e, DD,
        agg, nullptr, nullptr, DD);
    gemm_ws<1, 0><<<dim3(EE / 256, DLP / 64), 256, 0, stream>>>(
        w2vb, nullptr, nullptr, WtEL, KL, edge_src, edge_dst, b_el, 300,
        aggl, nullptr, nullptr, DLP);

    k_norm<<<cdiv(NN * DD, 256), 256, 0, stream>>>(agg, cnt, aggb, DD);
    k_norm<<<cdiv(NN * DLP, 256), 256, 0, stream>>>(aggl, cnt, agglb, DLP);

    gemm_ws<2, 1><<<dim3(NN / 256, DD / 64), 256, 0, stream>>>(
        featb, aggb, nullptr, WtNU, KNU, nullptr, nullptr, b_nu, DD,
        nullptr, nnfb, nullptr, DD);
    gemm_ws<3, 1><<<dim3(NN / 256, DLP / 64), 256, 0, stream>>>(
        w2vb, agglb, nullptr, WtNUL, KNL, nullptr, nullptr, b_nul, 300,
        nullptr, nnflb, nullptr, DLP);

    gemm_ws<4, 2><<<dim3(ERO / 256, CPAD / 64), 256, 0, stream>>>(
        nnfb, nnflb, sfrob, WtP, KP, ro_src, ro_dst, b_p, CC,
        nullptr, nullptr, out, CC);
}
